// BiUnet_3951369912850
// MI455X (gfx1250) — matrix operand order repaired and checked by reading, not compiled, not run on hardware
//
#include <hip/hip_runtime.h>
#include <hip/hip_bf16.h>

typedef _Float16 v16h __attribute__((ext_vector_type(16)));
typedef _Float16 v8h  __attribute__((ext_vector_type(8)));
typedef float    v8f  __attribute__((ext_vector_type(8)));
typedef float    v4f  __attribute__((ext_vector_type(4)));

#define BB      2
#define NHEAD   8
#define CDIM    256
#define HD      32
#define HWDIM   56
#define S_IMG   3136
#define NQTOK   15
#define STOT    3151
#define SPAD    3168          /* pad to multiple of 32 keys */
#define MWIN    50
#define QTILES  197
#define KCHUNKS 99            /* ceil(197/2): 32-key chunks */
#define RROWS   (BB * STOT)   /* 6302 */
#define ROWT    394           /* ceil(6302/16) */

static __device__ __forceinline__ v8f wmma_f16(v16h a, v16h b, v8f c) {
    return __builtin_amdgcn_wmma_f32_16x16x32_f16(false, a, false, b, (short)0, c,
                                                  false, false);
}
// identical back-to-back WMMA sharing the A operand: set reuse-A hint
static __device__ __forceinline__ v8f wmma_f16_ra(v16h a, v16h b, v8f c) {
    return __builtin_amdgcn_wmma_f32_16x16x32_f16(false, a, false, b, (short)0, c,
                                                  true, false);
}

static __device__ __forceinline__ v16h cat8(v8h lo, v8h hi) {
    v16h r;
#pragma unroll
    for (int i = 0; i < 8; ++i) { r[i] = lo[i]; r[i + 8] = hi[i]; }
    return r;
}

static __device__ __forceinline__ v8h ldcvt8(const float* p) {
    v4f a = *(const v4f*)p;
    v4f b = *(const v4f*)(p + 4);
    v8h r;
#pragma unroll
    for (int i = 0; i < 4; ++i) { r[i] = (_Float16)a[i]; r[i + 4] = (_Float16)b[i]; }
    return r;
}

// B fragment (32x16 f16): col = lane&15, halves = K = 8*hv + i, 16 + 8*hv + i  (table order)
static __device__ __forceinline__ v16h ldbf(const _Float16* wcol, int kk, int hv) {
    return cat8(*(const v8h*)(wcol + kk * 32 + 8 * hv),
                *(const v8h*)(wcol + kk * 32 + 16 + 8 * hv));
}

// ---------------------------------------------------------------------------
// Prep: transpose Wqkv [256,768] -> Wt_f16 [768][256], Wo [256,256] -> Wot_f16
// ---------------------------------------------------------------------------
__global__ void prep_weights_kernel(const float* __restrict__ Wqkv,
                                    const float* __restrict__ Wo,
                                    _Float16* __restrict__ Wt,
                                    _Float16* __restrict__ Wot) {
    int i = blockIdx.x * blockDim.x + threadIdx.x;
    if (i < 768 * 256) {
        int n = i >> 8, k = i & 255;
        Wt[i] = (_Float16)Wqkv[k * 768 + n];
    }
    if (i < 256 * 256) {
        int n = i >> 8, k = i & 255;
        Wot[i] = (_Float16)Wo[k * 256 + n];
    }
}

// ---------------------------------------------------------------------------
// QKV projection (6302 x 256 @ 256 x 768). One wave owns a 16-row strip and
// FOUR 16-wide column tiles: A fragment loaded once per K-step, 4 chained
// WMMAs with reuse-A. Results scattered into head-major windowed order (f16)
// plus f32 v image for the lepe conv.
// ---------------------------------------------------------------------------
__global__ void qkv_kernel(const float* __restrict__ x,
                           const float* __restrict__ cls,
                           const _Float16* __restrict__ Wt,
                           const float* __restrict__ bqkv,
                           _Float16* __restrict__ qh,
                           _Float16* __restrict__ kh,
                           _Float16* __restrict__ vh,
                           float* __restrict__ vimg) {
    const int wave = threadIdx.x >> 5, lane = threadIdx.x & 31;
    const int task = blockIdx.x * 4 + wave;
    const int CGRP = 12;  // 48 col tiles / 4
    if (task >= ROWT * CGRP) return;
    const int rt = task / CGRP, cg = task % CGRP;
    const int n = lane & 15, hv = lane >> 4;

    int rg = rt * 16 + n;
    if (rg >= RROWS) rg = RROWS - 1;
    const int b = rg / STOT, s = rg - b * STOT;
    const float* rowp = (s < S_IMG)
        ? (x + ((size_t)b * S_IMG + s) * CDIM)
        : (cls + ((size_t)b * NQTOK + (s - S_IMG)) * CDIM);

    const _Float16* w0 = Wt + (size_t)(cg * 64 + 0 * 16 + n) * CDIM;
    const _Float16* w1 = Wt + (size_t)(cg * 64 + 1 * 16 + n) * CDIM;
    const _Float16* w2 = Wt + (size_t)(cg * 64 + 2 * 16 + n) * CDIM;
    const _Float16* w3 = Wt + (size_t)(cg * 64 + 3 * 16 + n) * CDIM;

    v8f acc0, acc1, acc2, acc3;
#pragma unroll
    for (int i = 0; i < 8; ++i) { acc0[i] = 0.f; acc1[i] = 0.f; acc2[i] = 0.f; acc3[i] = 0.f; }

#pragma unroll
    for (int kk = 0; kk < 8; ++kk) {
        v16h a = cat8(ldcvt8(rowp + kk * 32 + 8 * hv),
                      ldcvt8(rowp + kk * 32 + 16 + 8 * hv));
        acc0 = wmma_f16   (a, ldbf(w0, kk, hv), acc0);
        acc1 = wmma_f16_ra(a, ldbf(w1, kk, hv), acc1);
        acc2 = wmma_f16_ra(a, ldbf(w2, kk, hv), acc2);
        acc3 = wmma_f16_ra(a, ldbf(w3, kk, hv), acc3);
    }

#pragma unroll
    for (int u = 0; u < 4; ++u) {
        const v8f acc = (u == 0) ? acc0 : (u == 1) ? acc1 : (u == 2) ? acc2 : acc3;
        const int c = cg * 64 + u * 16 + n;
        const int which = c >> 8, cc = c & 255, h = cc >> 5, d = cc & 31;
        const float bias = bqkv[c];
        _Float16* dst = (which == 0) ? qh : ((which == 1) ? kh : vh);
#pragma unroll
        for (int r = 0; r < 8; ++r) {
            int rg2 = rt * 16 + r + 8 * hv;
            if (rg2 >= RROWS) continue;
            int b2 = rg2 / STOT, s2 = rg2 - b2 * STOT;
            int sp;
            if (s2 < S_IMG) {
                int hh = s2 / HWDIM, ww = s2 - hh * HWDIM;
                sp = ((hh >> 3) * 7 + (ww >> 3)) * 64 + ((hh & 7) << 3) + (ww & 7);
            } else {
                sp = s2;
            }
            float val = acc[r] + bias;
            dst[((size_t)(b2 * NHEAD + h) * SPAD + sp) * HD + d] = (_Float16)val;
            if (which == 2 && s2 < S_IMG)
                vimg[((size_t)b2 * S_IMG + s2) * CDIM + cc] = val;
        }
    }
}

// ---------------------------------------------------------------------------
// lepe: depthwise 3x3 conv (SAME, groups=C) over f32 v image + bias
// ---------------------------------------------------------------------------
__global__ void lepe_kernel(const float* __restrict__ vimg,
                            const float* __restrict__ lw,
                            const float* __restrict__ lb,
                            float* __restrict__ lepe) {
    int i = blockIdx.x * blockDim.x + threadIdx.x;
    if (i >= BB * S_IMG * CDIM) return;
    int c = i & 255;
    int pix = i >> 8;
    int b = pix / S_IMG, p = pix - b * S_IMG;
    int hh = p / HWDIM, ww = p - hh * HWDIM;
    float acc = lb[c];
#pragma unroll
    for (int dh = 0; dh < 3; ++dh) {
        int y = hh + dh - 1;
        if ((unsigned)y >= HWDIM) continue;
#pragma unroll
        for (int dw = 0; dw < 3; ++dw) {
            int xx = ww + dw - 1;
            if ((unsigned)xx >= HWDIM) continue;
            acc += vimg[(((size_t)b * HWDIM + y) * HWDIM + xx) * CDIM + c] *
                   lw[(dh * 3 + dw) * CDIM + c];
        }
    }
    lepe[i] = acc;
}

// ---------------------------------------------------------------------------
// Block-sparse flash attention, 32-key chunks. One wave per (b,head,q-tile).
// Per unmasked chunk: V rows (2KB, contiguous) stream global->LDS with
// GLOBAL_LOAD_ASYNC_TO_LDS_B128 (overlapped with 2 logits WMMAs + online
// softmax), then s_wait_asynccnt drains, LDS transpose builds V^T, and two
// full-K P@V WMMAs accumulate. Mask byte gates the whole chunk.
// ---------------------------------------------------------------------------
__global__ void attn_kernel(const _Float16* __restrict__ qh,
                            const _Float16* __restrict__ kh,
                            const _Float16* __restrict__ vh,
                            const unsigned char* __restrict__ mask,
                            float* __restrict__ att) {
    __shared__ __align__(16) _Float16 ldsP[4][16 * 32];    // P tile (16q x 32k)
    __shared__ __align__(16) _Float16 ldsVt[4][32 * 32];   // V^T (d x key)
    __shared__ __align__(16) _Float16 ldsRaw[4][32 * 32];  // async V landing (row-major)

    const int wave = threadIdx.x >> 5, lane = threadIdx.x & 31;
    const int task = blockIdx.x * 4 + wave;
    if (task >= BB * NHEAD * QTILES) return;
    const int qt = task % QTILES;
    const int bh = task / QTILES;
    const int b = bh / NHEAD;
    const int n = lane & 15, hv = lane >> 4;

    const _Float16* qb = qh + (size_t)bh * SPAD * HD;
    const _Float16* kb = kh + (size_t)bh * SPAD * HD;
    const _Float16* vb = vh + (size_t)bh * SPAD * HD;

    const int qrow = qt * 16 + n;
    const v16h aQ = cat8(*(const v8h*)(qb + qrow * HD + 16 * hv),
                         *(const v8h*)(qb + qrow * HD + 16 * hv + 8));

    const int qwin = (qt * 16) >> 6;
    const unsigned char* mrow = mask + ((size_t)b * MWIN + qwin) * MWIN;

    float mr[8], lr[8];
    v8f o0, o1;
#pragma unroll
    for (int r = 0; r < 8; ++r) { mr[r] = -3.0e38f; lr[r] = 0.f; o0[r] = 0.f; o1[r] = 0.f; }

    const float sc = 0.0625f * 1.44269504f;  // C^-0.5 * log2(e)
    _Float16* lp  = ldsP[wave];
    _Float16* lvt = ldsVt[wave];
    const unsigned rawbase = (unsigned)(unsigned long long)&ldsRaw[wave][0];

    for (int j2 = 0; j2 < KCHUNKS; ++j2) {
        const int kwin = (j2 * 32) >> 6;
        if (!mrow[kwin]) continue;  // whole 32-key chunk masked off
        const int kbase = j2 * 32;

        // --- async stage V chunk (32 rows x 64B, contiguous 2KB) into LDS ---
        {
            const char* gsrc = (const char*)(vb + (size_t)kbase * HD);
#pragma unroll
            for (int q = 0; q < 4; ++q) {
                const char* gp = gsrc + q * 512 + lane * 16;
                unsigned lo = rawbase + (unsigned)(q * 512 + lane * 16);
                asm volatile("global_load_async_to_lds_b128 %0, %1, off"
                             :: "v"(lo), "v"(gp) : "memory");
            }
        }
        __builtin_prefetch(kb + (size_t)(kbase + 32 < SPAD ? kbase + 32 : kbase) * HD, 0, 1);

        // --- logits for the two 16-key tiles of this chunk ---
        v8f sa0, sa1;
#pragma unroll
        for (int r = 0; r < 8; ++r) { sa0[r] = 0.f; sa1[r] = 0.f; }
        {
            const int kr0 = kbase + n, kr1 = kbase + 16 + n;
            v16h bK0 = cat8(*(const v8h*)(kb + kr0 * HD + 16 * hv),
                            *(const v8h*)(kb + kr0 * HD + 16 * hv + 8));
            v16h bK1 = cat8(*(const v8h*)(kb + kr1 * HD + 16 * hv),
                            *(const v8h*)(kb + kr1 * HD + 16 * hv + 8));
            sa0 = wmma_f16   (aQ, bK0, sa0);
            sa1 = wmma_f16_ra(aQ, bK1, sa1);
        }

        const bool ki0 = (kbase + n) >= STOT;
        const bool ki1 = (kbase + 16 + n) >= STOT;
        float sv0[8], sv1[8], alpha[8], p0[8], p1[8];
#pragma unroll
        for (int r = 0; r < 8; ++r) {
            sv0[r] = ki0 ? -1.0e9f : sa0[r] * sc;
            sv1[r] = ki1 ? -1.0e9f : sa1[r] * sc;
        }
        // online softmax across the 32-key chunk (log2 domain)
#pragma unroll
        for (int r = 0; r < 8; ++r) {
            float v = fmaxf(sv0[r], sv1[r]);
            v = fmaxf(v, __shfl_xor(v, 1));
            v = fmaxf(v, __shfl_xor(v, 2));
            v = fmaxf(v, __shfl_xor(v, 4));
            v = fmaxf(v, __shfl_xor(v, 8));
            float mn = fmaxf(mr[r], v);
            alpha[r] = exp2f(mr[r] - mn);
            mr[r] = mn;
            p0[r] = exp2f(sv0[r] - mn);
            p1[r] = exp2f(sv1[r] - mn);
            float rs = p0[r] + p1[r];
            rs += __shfl_xor(rs, 1);
            rs += __shfl_xor(rs, 2);
            rs += __shfl_xor(rs, 4);
            rs += __shfl_xor(rs, 8);
            lr[r] = lr[r] * alpha[r] + rs;
        }
        // P (D layout -> A layout 16x32) through LDS
#pragma unroll
        for (int r = 0; r < 8; ++r) {
            lp[(r + 8 * hv) * 32 + n]      = (_Float16)p0[r];
            lp[(r + 8 * hv) * 32 + 16 + n] = (_Float16)p1[r];
        }

        // --- drain async V, transpose row-major -> V^T[d][key] in LDS ---
        asm volatile("s_wait_asynccnt 0" ::: "memory");
        {
            const _Float16* rw = ldsRaw[wave] + lane * HD;  // key row = lane
            v8h r0 = *(const v8h*)(rw);
            v8h r1 = *(const v8h*)(rw + 8);
            v8h r2 = *(const v8h*)(rw + 16);
            v8h r3 = *(const v8h*)(rw + 24);
#pragma unroll
            for (int i = 0; i < 8; ++i) {
                lvt[(i)      * 32 + lane] = r0[i];
                lvt[(8 + i)  * 32 + lane] = r1[i];
                lvt[(16 + i) * 32 + lane] = r2[i];
                lvt[(24 + i) * 32 + lane] = r3[i];
            }
        }
        asm volatile("s_wait_dscnt 0" ::: "memory");  // wave-local LDS ordering

        // fragments: P as A (full K=32), V^T columns as B (d groups 0/1)
        v16h aP  = cat8(*(const v8h*)(lp + n * 32 + 16 * hv),
                        *(const v8h*)(lp + n * 32 + 16 * hv + 8));
        v16h bV0 = cat8(*(const v8h*)(lvt + n * 32 + 16 * hv),
                        *(const v8h*)(lvt + n * 32 + 16 * hv + 8));
        v16h bV1 = cat8(*(const v8h*)(lvt + (16 + n) * 32 + 16 * hv),
                        *(const v8h*)(lvt + (16 + n) * 32 + 16 * hv + 8));
#pragma unroll
        for (int r = 0; r < 8; ++r) { o0[r] *= alpha[r]; o1[r] *= alpha[r]; }
        o0 = wmma_f16   (aP, bV0, o0);
        o1 = wmma_f16_ra(aP, bV1, o1);
    }

    float* ab = att + (size_t)bh * SPAD * HD;
#pragma unroll
    for (int r = 0; r < 8; ++r) {
        int q = qt * 16 + r + 8 * hv;
        if (q < STOT) {
            float inv = 1.0f / lr[r];
            ab[(size_t)q * HD + n]      = o0[r] * inv;
            ab[(size_t)q * HD + 16 + n] = o1[r] * inv;
        }
    }
}

// ---------------------------------------------------------------------------
// Output projection: y = (gather(att) [+ lepe]) @ Wo + bo. One wave owns a
// 16-row strip and all four of its 16-wide column-tile group: expensive A
// gather (att + lepe, permuted) amortized over 4 chained reuse-A WMMAs.
// ---------------------------------------------------------------------------
__global__ void proj_kernel(const float* __restrict__ att,
                            const float* __restrict__ lepe,
                            const _Float16* __restrict__ Wot,
                            const float* __restrict__ bo,
                            float* __restrict__ out) {
    const int wave = threadIdx.x >> 5, lane = threadIdx.x & 31;
    const int task = blockIdx.x * 4 + wave;
    const int CGRP = 4;  // 16 col tiles / 4
    if (task >= ROWT * CGRP) return;
    const int rt = task / CGRP, cg = task % CGRP;
    const int n = lane & 15, hv = lane >> 4;

    int rg = rt * 16 + n;
    if (rg >= RROWS) rg = RROWS - 1;
    const int b = rg / STOT, s = rg - b * STOT;
    const bool isimg = s < S_IMG;
    int sp;
    if (isimg) {
        int hh = s / HWDIM, ww = s - hh * HWDIM;
        sp = ((hh >> 3) * 7 + (ww >> 3)) * 64 + ((hh & 7) << 3) + (ww & 7);
    } else {
        sp = s;
    }
    const float* lep = lepe + ((size_t)b * S_IMG + (isimg ? s : 0)) * CDIM;
    const _Float16* w0 = Wot + (size_t)(cg * 64 + 0 * 16 + n) * CDIM;
    const _Float16* w1 = Wot + (size_t)(cg * 64 + 1 * 16 + n) * CDIM;
    const _Float16* w2 = Wot + (size_t)(cg * 64 + 2 * 16 + n) * CDIM;
    const _Float16* w3 = Wot + (size_t)(cg * 64 + 3 * 16 + n) * CDIM;

    v8f acc0, acc1, acc2, acc3;
#pragma unroll
    for (int i = 0; i < 8; ++i) { acc0[i] = 0.f; acc1[i] = 0.f; acc2[i] = 0.f; acc3[i] = 0.f; }

#pragma unroll
    for (int kk = 0; kk < 8; ++kk) {  // head index == K-step (heads are 32 wide)
        const float* ap = att + ((size_t)(b * NHEAD + kk) * SPAD + sp) * HD;
        v8h lo, hi;
#pragma unroll
        for (int i = 0; i < 8; ++i) {
            float v0 = ap[8 * hv + i];
            float v1 = ap[16 + 8 * hv + i];
            if (isimg) {
                v0 += lep[kk * 32 + 8 * hv + i];
                v1 += lep[kk * 32 + 16 + 8 * hv + i];
            }
            lo[i] = (_Float16)v0;
            hi[i] = (_Float16)v1;
        }
        v16h a = cat8(lo, hi);
        acc0 = wmma_f16   (a, ldbf(w0, kk, hv), acc0);
        acc1 = wmma_f16_ra(a, ldbf(w1, kk, hv), acc1);
        acc2 = wmma_f16_ra(a, ldbf(w2, kk, hv), acc2);
        acc3 = wmma_f16_ra(a, ldbf(w3, kk, hv), acc3);
    }

#pragma unroll
    for (int u = 0; u < 4; ++u) {
        const v8f acc = (u == 0) ? acc0 : (u == 1) ? acc1 : (u == 2) ? acc2 : acc3;
        const int c = cg * 64 + u * 16 + n;
        const float bias = bo[c];
#pragma unroll
        for (int r = 0; r < 8; ++r) {
            int rg2 = rt * 16 + r + 8 * hv;
            if (rg2 >= RROWS) continue;
            int b2 = rg2 / STOT, s2 = rg2 - b2 * STOT;
            float val = acc[r] + bias;
            if (s2 < S_IMG)
                out[((size_t)b2 * S_IMG + s2) * CDIM + c] = val;
            else
                out[(size_t)BB * S_IMG * CDIM +
                    ((size_t)b2 * NQTOK + (s2 - S_IMG)) * CDIM + c] = val;
        }
    }
}

// ---------------------------------------------------------------------------
extern "C" void kernel_launch(void* const* d_in, const int* in_sizes, int n_in,
                              void* d_out, int out_size, void* d_ws, size_t ws_size,
                              hipStream_t stream) {
    const float* x    = (const float*)d_in[0];
    const float* cls  = (const float*)d_in[1];
    const unsigned char* mask = (const unsigned char*)d_in[2];
    const float* Wqkv = (const float*)d_in[3];
    const float* bqkv = (const float*)d_in[4];
    const float* lw   = (const float*)d_in[5];
    const float* lb   = (const float*)d_in[6];
    const float* Wo   = (const float*)d_in[7];
    const float* bo   = (const float*)d_in[8];
    float* out = (float*)d_out;

    char* w = (char*)d_ws;
    size_t off = 0;
    auto alloc = [&](size_t bytes) -> void* {
        void* p = w + off;
        off = (off + bytes + 255) & ~(size_t)255;
        return p;
    };
    const size_t HSZ = (size_t)BB * NHEAD * SPAD * HD;  // per q/k/v/att
    _Float16* qh   = (_Float16*)alloc(HSZ * 2);
    _Float16* kh   = (_Float16*)alloc(HSZ * 2);
    _Float16* vh   = (_Float16*)alloc(HSZ * 2);
    float*    vimg = (float*)alloc((size_t)BB * S_IMG * CDIM * 4);
    float*    lepe = (float*)alloc((size_t)BB * S_IMG * CDIM * 4);
    float*    att  = (float*)alloc(HSZ * 4);
    _Float16* Wt   = (_Float16*)alloc((size_t)768 * 256 * 2);
    _Float16* Wot  = (_Float16*)alloc((size_t)256 * 256 * 2);

    prep_weights_kernel<<<768, 256, 0, stream>>>(Wqkv, Wo, Wt, Wot);

    qkv_kernel<<<(ROWT * 12) / 4, 128, 0, stream>>>(x, cls, Wt, bqkv, qh, kh, vh, vimg);

    lepe_kernel<<<(BB * S_IMG * CDIM) / 256, 256, 0, stream>>>(vimg, lw, lb, lepe);

    attn_kernel<<<(BB * NHEAD * QTILES + 3) / 4, 128, 0, stream>>>(qh, kh, vh, mask, att);

    proj_kernel<<<(ROWT * 4) / 4, 128, 0, stream>>>(att, lepe, Wot, bo, out);
}
